// Layer_6270652252843
// MI455X (gfx1250) — matrix operand order repaired and checked by reading, not compiled, not run on hardware
//
#include <hip/hip_runtime.h>

// ---------------- CDNA5 WMMA types ----------------
typedef __attribute__((ext_vector_type(16))) __bf16 v16bf;
typedef __attribute__((ext_vector_type(2)))  __bf16 v2bf;
typedef __attribute__((ext_vector_type(8)))  float  v8f;
typedef __attribute__((ext_vector_type(2)))  float  v2f;

union Frag {
    v16bf v;
    unsigned short h[16];
    unsigned u[8];
    uint4 q[2];
};

__device__ __forceinline__ unsigned short f2bf(float f) {
    return __builtin_bit_cast(unsigned short, (__bf16)f);
}
// one v_cvt_pk_bf16_f32: convert a pair and return packed u32
__device__ __forceinline__ unsigned pk2bf(float lo, float hi) {
    v2f t; t.x = lo; t.y = hi;
    v2bf b = __builtin_convertvector(t, v2bf);
    return __builtin_bit_cast(unsigned, b);
}

static constexpr int NB   = 16384;  // batch
static constexpr int NCOL = 4096;   // lattice sites per row
static constexpr int NH   = 64;     // hidden
static constexpr int HALF = 2048;   // sites per color

// ---------------- prep: f32 (R x C) -> bf16 transposed (C x R) ----------------
__global__ void transpose_bf16(const float* __restrict__ in, unsigned short* __restrict__ out,
                               int R, int C) {
    int i = blockIdx.x * blockDim.x + threadIdx.x;
    if (i >= R * C) return;
    int c = i / R, r = i % R;
    out[i] = f2bf(in[(size_t)r * C + c]);     // out[c*R + r]
}

// ---------------- fused coupling-layer kernel ----------------
__global__ __launch_bounds__(128) void coupling_kernel(
    const float* __restrict__ phi,
    const unsigned short* __restrict__ w1s, const unsigned short* __restrict__ w2s,
    const unsigned short* __restrict__ w3s,
    const unsigned short* __restrict__ w1t, const unsigned short* __restrict__ w2t,
    const unsigned short* __restrict__ w3t,
    const float* __restrict__ b1s, const float* __restrict__ b2s, const float* __restrict__ b3s,
    const float* __restrict__ b1t, const float* __restrict__ b2t, const float* __restrict__ b3t,
    float* __restrict__ z)
{
    __shared__ unsigned short lh1[2][4][16][64];   // [branch][wave][row][col] bf16, 16 KB
    __shared__ unsigned short lh2[2][4][16][64];   // 16 KB

    const int tid  = threadIdx.x;
    const int wv   = tid >> 5;      // wave in block (0..3)
    const int lane = tid & 31;
    const int n    = lane & 15;     // N index / A-row index
    const int hh   = lane >> 4;     // lane-half (K split for A/B fragments)
    const int row0 = blockIdx.x * 64 + wv * 16;

    // -------- Layer 1: x(16x2048) @ W1(2048x64), both branches --------
    v8f accS[4], accT[4];
#pragma unroll
    for (int c = 0; c < 4; ++c) { accS[c] = (v8f)(0.0f); accT[c] = (v8f)(0.0f); }

    auto l1step = [&](int kt, int off) {
        // K-tile kt covers phi cols [kt*64, kt*64+63]; even sites at 2k+off.
        // A frag (16x32 bf16): lane-half hh -> K = hh*8+{0..7} and 16+hh*8+{0..7}
        const float* pr = phi + (size_t)(row0 + n) * NCOL + kt * 64 + hh * 16;
        float f0[16], f1[16];
#pragma unroll
        for (int i = 0; i < 4; ++i) {
            *(float4*)&f0[4 * i] = ((const float4*)pr)[i];
            *(float4*)&f1[4 * i] = ((const float4*)(pr + 32))[i];
        }
        Frag a;
#pragma unroll
        for (int v = 0; v < 4; ++v) {
            a.u[v]     = pk2bf(f0[4 * v + off], f0[4 * v + 2 + off]);   // K = hh*8+2v, +1
            a.u[4 + v] = pk2bf(f1[4 * v + off], f1[4 * v + 2 + off]);   // K = 16+hh*8+2v, +1
        }
#pragma unroll
        for (int c = 0; c < 4; ++c) {
            // B frag (32x16): lane n = column, K = kt*32 + hh*8 + {0..7} and kt*32 + 16 + hh*8 + {0..7} in W1T
            const uint4* ps = (const uint4*)(w1s + (size_t)(c * 16 + n) * HALF + kt * 32 + hh * 8);
            const uint4* pt = (const uint4*)(w1t + (size_t)(c * 16 + n) * HALF + kt * 32 + hh * 8);
            Frag bS, bT;
            bS.q[0] = ps[0]; bS.q[1] = ps[2];
            bT.q[0] = pt[0]; bT.q[1] = pt[2];
            accS[c] = __builtin_amdgcn_wmma_f32_16x16x32_bf16(false, a.v, false, bS.v, (short)0, accS[c], false, false);
            accT[c] = __builtin_amdgcn_wmma_f32_16x16x32_bf16(false, a.v, false, bT.v, (short)0, accT[c], false, false);
        }
    };
#pragma unroll 1
    for (int k2 = 0; k2 < 32; ++k2) { l1step(2 * k2, 0); l1step(2 * k2 + 1, 1); }

    // bias + relu, D-layout (row = p + 8*hh, col = c*16+n) -> LDS as bf16
#pragma unroll
    for (int c = 0; c < 4; ++c) {
        float bbs = b1s[c * 16 + n], bbt = b1t[c * 16 + n];
#pragma unroll
        for (int p = 0; p < 8; ++p) {
            float vs = fmaxf(accS[c][p] + bbs, 0.0f);
            float vt = fmaxf(accT[c][p] + bbt, 0.0f);
            lh1[0][wv][p + 8 * hh][c * 16 + n] = f2bf(vs);
            lh1[1][wv][p + 8 * hh][c * 16 + n] = f2bf(vt);
        }
    }
    __syncthreads();

    // -------- Layer 2: h1(16x64) @ W2(64x64) --------
    Frag a2S[2], a2T[2];
#pragma unroll
    for (int ks = 0; ks < 2; ++ks) {
        a2S[ks].q[0] = *(const uint4*)&lh1[0][wv][n][ks * 32 + hh * 8];
        a2S[ks].q[1] = *(const uint4*)&lh1[0][wv][n][ks * 32 + 16 + hh * 8];
        a2T[ks].q[0] = *(const uint4*)&lh1[1][wv][n][ks * 32 + hh * 8];
        a2T[ks].q[1] = *(const uint4*)&lh1[1][wv][n][ks * 32 + 16 + hh * 8];
    }
    v8f h2S[4], h2T[4];
#pragma unroll
    for (int c = 0; c < 4; ++c) { h2S[c] = (v8f)(0.0f); h2T[c] = (v8f)(0.0f); }
#pragma unroll
    for (int c = 0; c < 4; ++c) {
#pragma unroll
        for (int ks = 0; ks < 2; ++ks) {
            const uint4* ps = (const uint4*)(w2s + (size_t)(c * 16 + n) * NH + ks * 32 + hh * 8);
            const uint4* pt = (const uint4*)(w2t + (size_t)(c * 16 + n) * NH + ks * 32 + hh * 8);
            Frag bS, bT;
            bS.q[0] = ps[0]; bS.q[1] = ps[2];
            bT.q[0] = pt[0]; bT.q[1] = pt[2];
            h2S[c] = __builtin_amdgcn_wmma_f32_16x16x32_bf16(false, a2S[ks].v, false, bS.v, (short)0, h2S[c], false, false);
            h2T[c] = __builtin_amdgcn_wmma_f32_16x16x32_bf16(false, a2T[ks].v, false, bT.v, (short)0, h2T[c], false, false);
        }
    }
#pragma unroll
    for (int c = 0; c < 4; ++c) {
        float bbs = b2s[c * 16 + n], bbt = b2t[c * 16 + n];
#pragma unroll
        for (int p = 0; p < 8; ++p) {
            float vs = fmaxf(h2S[c][p] + bbs, 0.0f);
            float vt = fmaxf(h2T[c][p] + bbt, 0.0f);
            lh2[0][wv][p + 8 * hh][c * 16 + n] = f2bf(vs);
            lh2[1][wv][p + 8 * hh][c * 16 + n] = f2bf(vt);
        }
    }
    __syncthreads();

    // -------- Layer 3: h2(16x64) @ W3(64x2048), fused epilogue --------
    Frag a3S[2], a3T[2];
#pragma unroll
    for (int ks = 0; ks < 2; ++ks) {
        a3S[ks].q[0] = *(const uint4*)&lh2[0][wv][n][ks * 32 + hh * 8];
        a3S[ks].q[1] = *(const uint4*)&lh2[0][wv][n][ks * 32 + 16 + hh * 8];
        a3T[ks].q[0] = *(const uint4*)&lh2[1][wv][n][ks * 32 + hh * 8];
        a3T[ks].q[1] = *(const uint4*)&lh2[1][wv][n][ks * 32 + 16 + hh * 8];
    }

#pragma unroll 1
    for (int ct = 0; ct < 128; ++ct) {
        const int j = ct * 16 + n;            // s/t column index (0..2047)
        v8f sAcc = (v8f)(0.0f), tAcc = (v8f)(0.0f);
#pragma unroll
        for (int ks = 0; ks < 2; ++ks) {
            const uint4* ps = (const uint4*)(w3s + (size_t)j * NH + ks * 32 + hh * 8);
            const uint4* pt = (const uint4*)(w3t + (size_t)j * NH + ks * 32 + hh * 8);
            Frag bS, bT;
            bS.q[0] = ps[0]; bS.q[1] = ps[2];
            bT.q[0] = pt[0]; bT.q[1] = pt[2];
            sAcc = __builtin_amdgcn_wmma_f32_16x16x32_bf16(false, a3S[ks].v, false, bS.v, (short)0, sAcc, false, false);
            tAcc = __builtin_amdgcn_wmma_f32_16x16x32_bf16(false, a3T[ks].v, false, bT.v, (short)0, tAcc, false, false);
        }
        const float sb = b3s[j], tb = b3t[j];
        const int rr = j >> 5, q = j & 31;
        const int colb = rr * 64 + 2 * q;     // adjacent (even,odd) site pair
        const int swap = rr & 1;              // odd lattice row: (odd,even) order
        const float* pb = phi + (size_t)(row0 + 8 * hh) * NCOL + colb;
        float*       zb = z   + (size_t)(row0 + 8 * hh) * NCOL + colb;
#pragma unroll
        for (int p = 0; p < 8; ++p) {
            float2 pv = *(const float2*)(pb + (size_t)p * NCOL);
            float s = sAcc[p] + sb;
            float t = tAcc[p] + tb;
            float es = __expf(-s);
            float ev = swap ? pv.y : pv.x;    // even (black) site: passthrough
            float ov = swap ? pv.x : pv.y;    // odd (white) site: transform
            float zo = (ov - t) * es;
            float2 r;
            r.x = swap ? zo : ev;
            r.y = swap ? ev : zo;
            *(float2*)(zb + (size_t)p * NCOL) = r;
        }
    }
}

// ---------------- launch ----------------
extern "C" void kernel_launch(void* const* d_in, const int* in_sizes, int n_in,
                              void* d_out, int out_size, void* d_ws, size_t ws_size,
                              hipStream_t stream) {
    const float* phi = (const float*)d_in[0];
    const float* Ws1 = (const float*)d_in[1];
    const float* bs1 = (const float*)d_in[2];
    const float* Ws2 = (const float*)d_in[3];
    const float* bs2 = (const float*)d_in[4];
    const float* Ws3 = (const float*)d_in[5];
    const float* bs3 = (const float*)d_in[6];
    const float* Wt1 = (const float*)d_in[7];
    const float* bt1 = (const float*)d_in[8];
    const float* Wt2 = (const float*)d_in[9];
    const float* bt2 = (const float*)d_in[10];
    const float* Wt3 = (const float*)d_in[11];
    const float* bt3 = (const float*)d_in[12];

    unsigned short* ws  = (unsigned short*)d_ws;
    unsigned short* w1s = ws + 0;       // W1T (64 x 2048) bf16
    unsigned short* w1t = ws + 131072;
    unsigned short* w2s = ws + 262144;  // W2T (64 x 64)
    unsigned short* w2t = ws + 266240;
    unsigned short* w3s = ws + 270336;  // W3T (2048 x 64)
    unsigned short* w3t = ws + 401408;

    transpose_bf16<<<(2048 * 64 + 255) / 256, 256, 0, stream>>>(Ws1, w1s, 2048, 64);
    transpose_bf16<<<(2048 * 64 + 255) / 256, 256, 0, stream>>>(Wt1, w1t, 2048, 64);
    transpose_bf16<<<(64 * 64 + 255) / 256, 256, 0, stream>>>(Ws2, w2s, 64, 64);
    transpose_bf16<<<(64 * 64 + 255) / 256, 256, 0, stream>>>(Wt2, w2t, 64, 64);
    transpose_bf16<<<(64 * 2048 + 255) / 256, 256, 0, stream>>>(Ws3, w3s, 64, 2048);
    transpose_bf16<<<(64 * 2048 + 255) / 256, 256, 0, stream>>>(Wt3, w3t, 64, 2048);

    coupling_kernel<<<NB / 64, 128, 0, stream>>>(
        phi, w1s, w2s, w3s, w1t, w2t, w3t,
        bs1, bs2, bs3, bt1, bt2, bt3, (float*)d_out);
}
